// ECA_11029476016517
// MI455X (gfx1250) — matrix operand order repaired and checked by reading, not compiled, not run on hardware
//
#include <hip/hip_runtime.h>
#include <hip/hip_bf16.h>
#include <math.h>

typedef __attribute__((ext_vector_type(16))) _Float16 v16h;
typedef __attribute__((ext_vector_type(8)))  float    v8f;
typedef __attribute__((ext_vector_type(4)))  float    v4f;

#define HW  128
#define IMG (HW * HW)          // 16384
#define BATCH 16
#define CHAN  128
#define CPB   8                // channels per block

union Frag16 { v16h h; uint4 u[2]; };

// A operand (16x32 f16, MxK), row-major source with ld=128.
// ISA 7.12.2: lanes 0-15 hold rows M=0..15 with K={0..7,16..23};
// lanes 16-31 hold same rows with K={8..15,24..31}.
__device__ __forceinline__ v16h load_a_frag(const _Float16* M, int m0, int k0, int lane) {
    Frag16 f;
    const int row = m0 + (lane & 15);
    const int kb  = k0 + ((lane >> 4) << 3);
    f.u[0] = *(const uint4*)(M + row * HW + kb);        // K = kb..kb+7
    f.u[1] = *(const uint4*)(M + row * HW + kb + 16);   // K = kb+16..kb+23
    return f.h;
}

// B operand (32x16 f16, KxN), row-major source with ld=128.
// Lane = N column (lane&15); lanes 0-15 hold K={0..7,16..23}, lanes 16-31 K={8..15,24..31}.
__device__ __forceinline__ v16h load_b_frag(const _Float16* M, int k0, int n0, int lane) {
    Frag16 f;
    const int col = n0 + (lane & 15);
    const int kb  = k0 + ((lane >> 4) << 3);
    #pragma unroll
    for (int s = 0; s < 8; ++s) {
        f.h[s]     = M[(kb + s) * HW + col];
        f.h[s + 8] = M[(kb + 16 + s) * HW + col];
    }
    return f.h;
}

__device__ __forceinline__ v8f wmma_f16(v16h a, v16h b, v8f c) {
    return __builtin_amdgcn_wmma_f32_16x16x32_f16(false, a, false, b, (short)0, c, false, false);
}

// ---------------- Kernel 1: build orthonormal DCT-II matrix (f16) ----------------
__global__ void gen_dct(_Float16* __restrict__ D, _Float16* __restrict__ DT) {
    int idx = blockIdx.x * blockDim.x + threadIdx.x;
    if (idx >= IMG) return;
    int k = idx >> 7, n = idx & 127;
    float scale = (k == 0) ? 0.08838834764831845f /* sqrt(1/128) */ : 0.125f /* sqrt(2/128) */;
    float v = cosf(3.14159265358979323846f * (2.0f * n + 1.0f) * (float)k / 256.0f) * scale;
    D [k * HW + n] = (_Float16)v;   // D  row-major
    DT[n * HW + k] = (_Float16)v;   // D^T row-major
}

// ---------------- Kernel 2: zero energy accumulator ----------------
__global__ void zero_f32(float* __restrict__ p, int n) {
    int i = blockIdx.x * blockDim.x + threadIdx.x;
    if (i < n) p[i] = 0.0f;
}

// ---------------- Kernel 3: 2D DCT (WMMA) + |.|/C energy, 8 channels per block ----------------
__global__ void __launch_bounds__(256) dct_energy(
    const float*    __restrict__ x,
    const _Float16* __restrict__ gD,
    const _Float16* __restrict__ gDT,
    float*          __restrict__ energy) {

    __shared__ _Float16 sD [IMG];  // D
    __shared__ _Float16 sDT[IMG];  // D^T
    __shared__ _Float16 sX [IMG];  // current channel slice (f16)
    __shared__ _Float16 sT [IMG];  // T = D @ X

    const int tid  = threadIdx.x;
    const int lane = tid & 31;
    const int wave = tid >> 5;                 // 0..7  -> owns tile column `wave`
    const int b    = blockIdx.x >> 4;          // 16 chunks per batch (CHAN/CPB)
    const int c0   = (blockIdx.x & 15) * CPB;

    // Stage D and D^T via CDNA5 async global->LDS copies (ASYNCcnt path).
    {
        const uint32_t dD  = (uint32_t)(uintptr_t)sD;
        const uint32_t dDT = (uint32_t)(uintptr_t)sDT;
        for (int i = tid; i < IMG / 8; i += 256) {
            const uint32_t o   = (uint32_t)i * 16u;
            const uint64_t srcD  = (uint64_t)(uintptr_t)(gD  + i * 8);
            const uint64_t srcDT = (uint64_t)(uintptr_t)(gDT + i * 8);
            asm volatile("global_load_async_to_lds_b128 %0, %1, off"
                         :: "v"(dD + o),  "v"(srcD)  : "memory");
            asm volatile("global_load_async_to_lds_b128 %0, %1, off"
                         :: "v"(dDT + o), "v"(srcDT) : "memory");
        }
        asm volatile("s_wait_asynccnt 0x0" ::: "memory");
    }
    __syncthreads();

    const int n0 = wave << 4;                  // this wave's output tile column

    // D^T B-fragments are invariant for the whole kernel: keep them in VGPRs.
    v16h bDT[4];
    #pragma unroll
    for (int s = 0; s < 4; ++s) bDT[s] = load_b_frag(sDT, s * 32, n0, lane);

    const v8f zero8 = {0.f, 0.f, 0.f, 0.f, 0.f, 0.f, 0.f, 0.f};
    v8f eacc[8];
    #pragma unroll
    for (int tt = 0; tt < 8; ++tt) eacc[tt] = zero8;

    for (int c = 0; c < CPB; ++c) {
        // Stage X(b, c0+c): f32 -> f16 into LDS
        const float* xim = x + (size_t)(b * CHAN + c0 + c) * IMG;
        for (int i = tid; i < IMG / 4; i += 256) {
            v4f v = ((const v4f*)xim)[i];
            _Float16* p = sX + i * 4;
            p[0] = (_Float16)v.x; p[1] = (_Float16)v.y;
            p[2] = (_Float16)v.z; p[3] = (_Float16)v.w;
        }
        __syncthreads();

        // Phase 2: T = D @ X. Wave computes tiles (tt, wave); B-frags loop-invariant.
        {
            v16h bX[4];
            #pragma unroll
            for (int s = 0; s < 4; ++s) bX[s] = load_b_frag(sX, s * 32, n0, lane);
            #pragma unroll
            for (int tt = 0; tt < 8; ++tt) {
                const int m0 = tt << 4;
                v8f acc = zero8;
                #pragma unroll
                for (int s = 0; s < 4; ++s)
                    acc = wmma_f16(load_a_frag(sD, m0, s * 32, lane), bX[s], acc);
                const int col = n0 + (lane & 15);
                const int rb  = m0 + ((lane >> 4) << 3);
                #pragma unroll
                for (int i = 0; i < 8; ++i)
                    sT[(rb + i) * HW + col] = (_Float16)acc[i];
            }
        }
        __syncthreads();

        // Phase 3: E = T @ D^T ; accumulate |E| into per-lane VGPR accumulators.
        #pragma unroll
        for (int tt = 0; tt < 8; ++tt) {
            const int m0 = tt << 4;
            v8f acc = zero8;
            #pragma unroll
            for (int s = 0; s < 4; ++s)
                acc = wmma_f16(load_a_frag(sT, m0, s * 32, lane), bDT[s], acc);
            #pragma unroll
            for (int i = 0; i < 8; ++i)
                eacc[tt][i] += fabsf(acc[i]);
        }
        __syncthreads();   // sT/sX reused next channel
    }

    // One atomic pass per block (8x fewer atomics than per-channel emission).
    float* eb = energy + (size_t)b * IMG;
    const float invC = 1.0f / (float)CHAN;
    const int col = n0 + (lane & 15);
    #pragma unroll
    for (int tt = 0; tt < 8; ++tt) {
        const int rb = (tt << 4) + ((lane >> 4) << 3);
        #pragma unroll
        for (int i = 0; i < 8; ++i)
            atomicAdd(&eb[(rb + i) * HW + col], eacc[tt][i] * invC);
    }
}

// ---------------- Kernel 4: top-16 extraction + MLP + sigmoid ----------------
__global__ void __launch_bounds__(256) topk_mlp(
    const float* __restrict__ energy,
    const float* __restrict__ w1, const float* __restrict__ b1,
    const float* __restrict__ w2, const float* __restrict__ b2,
    float* __restrict__ att) {

    __shared__ float se[IMG];
    __shared__ float rv[256];
    __shared__ int   ri[256];
    __shared__ float topv[16];

    const int b = blockIdx.x, tid = threadIdx.x;
    const float* eb = energy + (size_t)b * IMG;
    for (int i = tid; i < IMG; i += 256) se[i] = eb[i];
    __syncthreads();

    // 16 rounds of extract-max (descending order like lax.top_k)
    for (int s = 0; s < 16; ++s) {
        float mv = -3.4e38f; int mi = 0;
        for (int i = tid; i < IMG; i += 256) {
            float v = se[i];
            if (v > mv) { mv = v; mi = i; }
        }
        rv[tid] = mv; ri[tid] = mi;
        __syncthreads();
        if (tid == 0) {
            float bv = rv[0]; int bi = ri[0];
            for (int t = 1; t < 256; ++t)
                if (rv[t] > bv) { bv = rv[t]; bi = ri[t]; }
            topv[s] = bv;
            se[bi]  = -3.4e38f;
        }
        __syncthreads();
    }

    if (tid == 0) {
        float h[4];
        #pragma unroll
        for (int j = 0; j < 4; ++j) {
            float a = b1[j];
            #pragma unroll
            for (int i = 0; i < 16; ++i) a += topv[i] * w1[i * 4 + j];
            h[j] = a > 0.0f ? a : 0.0f;
        }
        float z = b2[0];
        #pragma unroll
        for (int j = 0; j < 4; ++j) z += h[j] * w2[j];
        att[b] = 1.0f / (1.0f + expf(-z));
    }
}

// ---------------- Kernel 5: out = x * att[b] ----------------
__global__ void scale_out(const float* __restrict__ x, const float* __restrict__ att,
                          float* __restrict__ out, long n4) {
    long i = (long)blockIdx.x * blockDim.x + threadIdx.x;
    if (i >= n4) return;
    int b = (int)((i * 4) >> 21);             // 128*128*128 = 2^21 elems per batch
    float a = att[b];
    v4f v = ((const v4f*)x)[i];
    v *= a;
    __builtin_nontemporal_store(v, (v4f*)out + i);   // streaming 128-bit store
}

extern "C" void kernel_launch(void* const* d_in, const int* in_sizes, int n_in,
                              void* d_out, int out_size, void* d_ws, size_t ws_size,
                              hipStream_t stream) {
    (void)in_sizes; (void)n_in; (void)out_size; (void)ws_size;

    const float* x  = (const float*)d_in[0];   // [16,128,128,128]
    const float* w1 = (const float*)d_in[1];   // [16,4]
    const float* b1 = (const float*)d_in[2];   // [4]
    const float* w2 = (const float*)d_in[3];   // [4,1]
    const float* b2 = (const float*)d_in[4];   // [1]
    float* out = (float*)d_out;

    // Workspace layout
    char* ws = (char*)d_ws;
    _Float16* wsD    = (_Float16*)(ws);                      // 32 KB
    _Float16* wsDT   = (_Float16*)(ws + 32768);              // 32 KB
    float*    energy = (float*)(ws + 65536);                 // 1 MB
    float*    att    = (float*)(ws + 65536 + BATCH * IMG * sizeof(float));

    gen_dct   <<<IMG / 256, 256, 0, stream>>>(wsD, wsDT);
    zero_f32  <<<(BATCH * IMG) / 256, 256, 0, stream>>>(energy, BATCH * IMG);
    dct_energy<<<BATCH * (CHAN / CPB), 256, 0, stream>>>(x, wsD, wsDT, energy);
    topk_mlp  <<<BATCH, 256, 0, stream>>>(energy, w1, b1, w2, b2, att);

    long n4 = (long)BATCH * CHAN * IMG / 4;                  // 8388608 float4
    scale_out <<<(int)(n4 / 256), 256, 0, stream>>>(x, att, out, n4);
}
